// LSTM_6786048327914
// MI455X (gfx1250) — matrix operand order repaired and checked by reading, not compiled, not run on hardware
//
#include <hip/hip_runtime.h>
#include <cstdint>
#include <cstddef>

typedef _Float16 half_t;
typedef __attribute__((ext_vector_type(16))) _Float16 v16h;
typedef __attribute__((ext_vector_type(8)))  _Float16 v8h;
typedef __attribute__((ext_vector_type(8)))  float    v8f;

#define N_BATCH 1024
#define F_IN    22
#define T_STEPS 1000
#define H_DIM   64
#define G_DIM   256   // 4*H
#define H_STRIDE 72   // hbuf row stride (halfs): keeps 16B alignment, spreads banks

// Branch-free hardware transcendentals (v_exp_f32 computes 2^x; v_rcp_f32).
#define LOG2E 1.44269504088896f
static __device__ __forceinline__ float fast_sigmoid(float x) {
    return __builtin_amdgcn_rcpf(1.0f + __builtin_amdgcn_exp2f(-LOG2E * x));
}
static __device__ __forceinline__ float fast_tanh(float x) {
    return 2.0f * __builtin_amdgcn_rcpf(1.0f + __builtin_amdgcn_exp2f(-2.0f * LOG2E * x)) - 1.0f;
}

// Workgroup barrier ordering only LDS traffic (s_wait_dscnt), NOT vector
// loads: the software-pipelined global x-load stays outstanding across it.
static __device__ __forceinline__ void wg_barrier_lds() {
    asm volatile(
        "s_wait_dscnt 0x0\n\t"
        "s_barrier_signal -1\n\t"
        "s_barrier_wait -1"
        ::: "memory");
}

// ---------------------------------------------------------------------------
// One-time data preparation kernels (f32 -> f16, layout fixes)
// ---------------------------------------------------------------------------

// x: (N, F, T) f32  ->  xT: (T, N, 32) f16, K padded 22->32 with zeros.
__global__ __launch_bounds__(256)
void convert_x_kernel(const float* __restrict__ x, half_t* __restrict__ xT) {
    int g = blockIdx.x * 256 + threadIdx.x;
    if (g >= N_BATCH * T_STEPS) return;
    int n = g / T_STEPS;
    int t = g % T_STEPS;              // consecutive lanes -> consecutive t -> coalesced reads
    half_t* dst = xT + ((size_t)t * N_BATCH + n) * 32;
#pragma unroll
    for (int k = 0; k < 32; ++k) {
        float v = (k < F_IN) ? x[((size_t)n * F_IN + k) * T_STEPS + t] : 0.0f;
        dst[k] = (half_t)v;
    }
}

// Generic (rows, ksrc) f32 -> (rows, kdst) f16 with zero K-padding.
__global__ __launch_bounds__(256)
void convert_pad_kernel(const float* __restrict__ src, half_t* __restrict__ dst,
                        int rows, int ksrc, int kdst) {
    int g = blockIdx.x * 256 + threadIdx.x;
    if (g >= rows * kdst) return;
    int r = g / kdst, k = g % kdst;
    dst[g] = (k < ksrc) ? (half_t)src[r * ksrc + k] : (half_t)0.0f;
}

__global__ __launch_bounds__(256)
void bias_sum_kernel(const float* __restrict__ a, const float* __restrict__ b,
                     float* __restrict__ o, int n) {
    int g = blockIdx.x * 256 + threadIdx.x;
    if (g < n) o[g] = a[g] + b[g];
}

// ---------------------------------------------------------------------------
// LSTM layer: persistent recurrence. One workgroup = one batch tile of M=16.
// 128 threads = 4 wave32. Wave q owns gate tiles {q, q+4, q+8, q+12}: one
// 16-column tile from EACH of i,f,g,o, so all four pre-activations for a cell
// (m, hc) sit in the same lane/element of the four accumulators -> the whole
// gate update runs in registers (no z exchange, no z barrier). h goes back to
// A-fragment layout through a double-buffered LDS tile: ONE barrier per step.
// KCH = number of 32-wide K chunks of the layer input (1 for layer 0, 2 else).
// ---------------------------------------------------------------------------

static __device__ __forceinline__ v16h combine8(v8h lo, v8h hi) {
    return __builtin_shufflevector(lo, hi, 0,1,2,3,4,5,6,7,8,9,10,11,12,13,14,15);
}

template <int KCH, bool WRITE_ALL>
__global__ __launch_bounds__(128)
void lstm_layer_kernel(const half_t* __restrict__ xin,   // [T][N][KCH*32] f16
                       const half_t* __restrict__ wih,   // [256][KCH*32] f16 (row n = gate row)
                       const half_t* __restrict__ whh,   // [256][64] f16
                       const float*  __restrict__ bias,  // [256] f32 (bih+bhh)
                       half_t*       __restrict__ hs_out)// [T][N][64] f16 (or [N][64] if !WRITE_ALL)
{
    constexpr int KIN = KCH * 32;

    __shared__ half_t hping[16 * H_STRIDE];  // h tiles, double buffered
    __shared__ half_t hpong[16 * H_STRIDE];

    const int tid  = threadIdx.x;
    const int q    = tid >> 5;                // wave = h-column group (0..3)
    const int lane = tid & 31;
    const int n0   = blockIdx.x * 16;         // batch rows [n0, n0+16)
    const int nn   = lane & 15;               // column within a 16-wide N tile
    const int khalf  = (lane < 16) ? 0 : 8;   // A-fragment K sub-select
    const int ksel16 = (lane < 16) ? 0 : 16;  // B-fragment K sub-select
    const int am     = lane & 15;             // A-fragment batch row (within tile)
    const int mbase  = (lane >> 4) << 3;      // C-fragment row offset (M = r + 8*(lane>=16))
    const int q16    = q * 16;

    // ---- Resident weights: B-fragments for tiles {4g+q} live in VGPRs ----
    v16h wih_r[4][KCH];
    v16h whh_r[4][2];
    float bb[4];
#pragma unroll
    for (int g = 0; g < 4; ++g) {
        const int row = (4 * g + q) * 16 + nn;        // gate-matrix row (0..255)
        bb[g] = bias[row];
#pragma unroll
        for (int kc = 0; kc < KCH; ++kc)
            wih_r[g][kc] = combine8(*(const v8h*)(wih + (size_t)row * KIN + kc * 32 + khalf),
                                    *(const v8h*)(wih + (size_t)row * KIN + kc * 32 + khalf + 16));
#pragma unroll
        for (int kc = 0; kc < 2; ++kc)
            whh_r[g][kc] = combine8(*(const v8h*)(whh + (size_t)row * H_DIM + kc * 32 + khalf),
                                    *(const v8h*)(whh + (size_t)row * H_DIM + kc * 32 + khalf + 16));
    }

    // ---- c state: 8 cells per lane, (m = mbase+r, hc = q16+nn) ----
    float creg[8];
#pragma unroll
    for (int r = 0; r < 8; ++r) creg[r] = 0.0f;
    for (int i = tid; i < 16 * H_STRIDE; i += 128) hping[i] = (half_t)0.0f;  // h_{-1} = 0

    const size_t xstride = (size_t)N_BATCH * KIN;          // halfs per timestep
    const half_t* xbase = xin + (size_t)(n0 + am) * KIN + khalf;
    // h output base: row (n0+mbase), column q16+nn; row r at immediate offset r*64.
    half_t* houtp = hs_out + (size_t)(n0 + mbase) * H_DIM + q16 + nn;

    // ---- prologue: load input A-fragments for t=0; xnext points at t=1 ----
    v16h a_pp[2][KCH];
#pragma unroll
    for (int kc = 0; kc < KCH; ++kc)
        a_pp[0][kc] = combine8(*(const v8h*)(xbase + kc * 32),
                               *(const v8h*)(xbase + kc * 32 + 16));
    const half_t* xnext = xbase + xstride;

    wg_barrier_lds();

    // One recurrence step: reads h_{t-1} from hread, writes h_t to hwrite.
    auto step = [&](int t, const v16h (&a_use)[KCH], v16h (&a_load)[KCH],
                    const half_t* hread, half_t* hwrite) {
        // pipelined load of t+1 input A-fragments (waited at next step's WMMA)
#pragma unroll
        for (int kc = 0; kc < KCH; ++kc)
            a_load[kc] = combine8(*(const v8h*)(xnext + kc * 32),
                                  *(const v8h*)(xnext + kc * 32 + 16));
        xnext += (t + 2 < T_STEPS) ? xstride : (size_t)0;

        // hidden A-fragments from LDS (previous step's buffer)
        v16h a_h[2];
#pragma unroll
        for (int kc = 0; kc < 2; ++kc) {
            const half_t* hrow = hread + am * H_STRIDE + kc * 32 + khalf;
            a_h[kc] = combine8(*(const v8h*)hrow, *(const v8h*)(hrow + 16));
        }

        // z_g = x_t @ Wih^T + h @ Whh^T + b for the 4 gate tiles of this wave
        v8f acc[4];
#pragma unroll
        for (int g = 0; g < 4; ++g) {
#pragma unroll
            for (int r = 0; r < 8; ++r) acc[g][r] = bb[g];
#pragma unroll
            for (int kc = 0; kc < KCH; ++kc)
                acc[g] = __builtin_amdgcn_wmma_f32_16x16x32_f16(false, a_use[kc], false, wih_r[g][kc],
                                                                (short)0, acc[g], false, false);
#pragma unroll
            for (int kc = 0; kc < 2; ++kc)
                acc[g] = __builtin_amdgcn_wmma_f32_16x16x32_f16(false, a_h[kc], false, whh_r[g][kc],
                                                                (short)0, acc[g], false, false);
        }

        // gates fully in registers: i=acc[0], f=acc[1], g=acc[2], o=acc[3]
#pragma unroll
        for (int r = 0; r < 8; ++r) {
            const float ig = fast_sigmoid(acc[0][r]);
            const float fg = fast_sigmoid(acc[1][r]);
            const float gg = fast_tanh(acc[2][r]);
            const float og = fast_sigmoid(acc[3][r]);
            const float c  = fg * creg[r] + ig * gg;
            creg[r] = c;
            const float h = og * fast_tanh(c);
            const half_t h16 = (half_t)h;
            hwrite[(mbase + r) * H_STRIDE + q16 + nn] = h16;   // transpose to A layout
            if constexpr (WRITE_ALL) {
                houtp[(size_t)r * H_DIM] = h16;
            } else {
                if (t == T_STEPS - 1) houtp[(size_t)r * H_DIM] = h16;
            }
        }
        if constexpr (WRITE_ALL) houtp += (size_t)N_BATCH * H_DIM;

        wg_barrier_lds();   // h_t visible; pipelined global x-load stays in flight
    };

    // T_STEPS is even: ping-pong registers AND LDS h buffers, no copies.
    for (int t = 0; t < T_STEPS; t += 2) {
        step(t,     a_pp[0], a_pp[1], hping, hpong);
        step(t + 1, a_pp[1], a_pp[0], hpong, hping);
    }
}

// ---------------------------------------------------------------------------
// MLP head: per-row block, 64 threads. 64->54->44->24->4 with BN+ReLU.
// ---------------------------------------------------------------------------
__global__ __launch_bounds__(64)
void head_kernel(const half_t* __restrict__ hfin,
                 const float* w0, const float* b0, const float* g0, const float* be0,
                 const float* m0, const float* v0,
                 const float* w1, const float* b1, const float* g1, const float* be1,
                 const float* m1, const float* v1,
                 const float* w2, const float* b2, const float* g2, const float* be2,
                 const float* m2, const float* v2,
                 const float* w3, const float* b3, float* __restrict__ out)
{
    __shared__ float s0[64], s1[54], s2[44], s3[24];
    const int row = blockIdx.x;
    const int j = threadIdx.x;
    s0[j] = (float)hfin[(size_t)row * 64 + j];
    __syncthreads();
    if (j < 54) {
        float a = b0[j];
        for (int k = 0; k < 64; ++k) a += w0[j * 64 + k] * s0[k];
        a = (a - m0[j]) * rsqrtf(v0[j] + 1e-5f) * g0[j] + be0[j];
        s1[j] = a > 0.0f ? a : 0.0f;
    }
    __syncthreads();
    if (j < 44) {
        float a = b1[j];
        for (int k = 0; k < 54; ++k) a += w1[j * 54 + k] * s1[k];
        a = (a - m1[j]) * rsqrtf(v1[j] + 1e-5f) * g1[j] + be1[j];
        s2[j] = a > 0.0f ? a : 0.0f;
    }
    __syncthreads();
    if (j < 24) {
        float a = b2[j];
        for (int k = 0; k < 44; ++k) a += w2[j * 44 + k] * s2[k];
        a = (a - m2[j]) * rsqrtf(v2[j] + 1e-5f) * g2[j] + be2[j];
        s3[j] = a > 0.0f ? a : 0.0f;
    }
    __syncthreads();
    if (j < 4) {
        float a = b3[j];
        for (int k = 0; k < 24; ++k) a += w3[j * 24 + k] * s3[k];
        out[(size_t)row * 4 + j] = a;
    }
}

// ---------------------------------------------------------------------------
// Launch
// ---------------------------------------------------------------------------
extern "C" void kernel_launch(void* const* d_in, const int* in_sizes, int n_in,
                              void* d_out, int out_size, void* d_ws, size_t ws_size,
                              hipStream_t stream) {
    (void)in_sizes; (void)n_in; (void)out_size; (void)ws_size;

    const float* x = (const float*)d_in[0];
    const float* Wih[3] = {(const float*)d_in[1], (const float*)d_in[5], (const float*)d_in[9]};
    const float* Whh[3] = {(const float*)d_in[2], (const float*)d_in[6], (const float*)d_in[10]};
    const float* bih[3] = {(const float*)d_in[3], (const float*)d_in[7], (const float*)d_in[11]};
    const float* bhh[3] = {(const float*)d_in[4], (const float*)d_in[8], (const float*)d_in[12]};
    const float* w0 = (const float*)d_in[13]; const float* b0 = (const float*)d_in[14];
    const float* w1 = (const float*)d_in[15]; const float* b1 = (const float*)d_in[16];
    const float* w2 = (const float*)d_in[17]; const float* b2 = (const float*)d_in[18];
    const float* w3 = (const float*)d_in[19]; const float* b3 = (const float*)d_in[20];
    const float* g0 = (const float*)d_in[21]; const float* be0 = (const float*)d_in[22];
    const float* m0 = (const float*)d_in[23]; const float* v0  = (const float*)d_in[24];
    const float* g1 = (const float*)d_in[25]; const float* be1 = (const float*)d_in[26];
    const float* m1 = (const float*)d_in[27]; const float* v1  = (const float*)d_in[28];
    const float* g2 = (const float*)d_in[29]; const float* be2 = (const float*)d_in[30];
    const float* m2 = (const float*)d_in[31]; const float* v2  = (const float*)d_in[32];
    float* out = (float*)d_out;

    // ---- workspace carve-up (256B aligned) ----
    uint8_t* ws = (uint8_t*)d_ws;
    size_t off = 0;
    auto alloc = [&](size_t bytes) -> uint8_t* {
        uint8_t* p = ws + off;
        off += (bytes + 255) & ~(size_t)255;
        return p;
    };
    half_t* xT   = (half_t*)alloc((size_t)T_STEPS * N_BATCH * 32 * sizeof(half_t));
    half_t* hs0  = (half_t*)alloc((size_t)T_STEPS * N_BATCH * H_DIM * sizeof(half_t));
    half_t* hs1  = (half_t*)alloc((size_t)T_STEPS * N_BATCH * H_DIM * sizeof(half_t));
    half_t* hfin = (half_t*)alloc((size_t)N_BATCH * H_DIM * sizeof(half_t));
    half_t* wih16[3]; half_t* whh16[3]; float* bsum[3];
    for (int l = 0; l < 3; ++l) {
        wih16[l] = (half_t*)alloc((size_t)G_DIM * 64 * sizeof(half_t));
        whh16[l] = (half_t*)alloc((size_t)G_DIM * 64 * sizeof(half_t));
        bsum[l]  = (float*)alloc((size_t)G_DIM * sizeof(float));
    }

    // ---- one-time prep ----
    {
        int tot = N_BATCH * T_STEPS;
        convert_x_kernel<<<(tot + 255) / 256, 256, 0, stream>>>(x, xT);
    }
    const int kin_src[3] = {F_IN, H_DIM, H_DIM};
    const int kin_dst[3] = {32, 64, 64};
    for (int l = 0; l < 3; ++l) {
        int tot_i = G_DIM * kin_dst[l];
        convert_pad_kernel<<<(tot_i + 255) / 256, 256, 0, stream>>>(Wih[l], wih16[l], G_DIM, kin_src[l], kin_dst[l]);
        int tot_h = G_DIM * H_DIM;
        convert_pad_kernel<<<(tot_h + 255) / 256, 256, 0, stream>>>(Whh[l], whh16[l], G_DIM, H_DIM, H_DIM);
        bias_sum_kernel<<<1, 256, 0, stream>>>(bih[l], bhh[l], bsum[l], G_DIM);
    }

    // ---- recurrence: 64 independent batch-tile workgroups per layer ----
    lstm_layer_kernel<1, true ><<<N_BATCH / 16, 128, 0, stream>>>(xT,  wih16[0], whh16[0], bsum[0], hs0);
    lstm_layer_kernel<2, true ><<<N_BATCH / 16, 128, 0, stream>>>(hs0, wih16[1], whh16[1], bsum[1], hs1);
    lstm_layer_kernel<2, false><<<N_BATCH / 16, 128, 0, stream>>>(hs1, wih16[2], whh16[2], bsum[2], hfin);

    // ---- head ----
    head_kernel<<<N_BATCH, 64, 0, stream>>>(hfin,
        w0, b0, g0, be0, m0, v0,
        w1, b1, g1, be1, m1, v1,
        w2, b2, g2, be2, m2, v2,
        w3, b3, out);
}
